// Encoder_13297218749026
// MI455X (gfx1250) — matrix operand order repaired and checked by reading, not compiled, not run on hardware
//
#include <hip/hip_runtime.h>
#include <hip/hip_bf16.h>

// GRU encoder for MI455X (gfx1250, wave32, WMMA bf16 16x16x32 + TDM loads).
// B=128, T=256, E=300 (pad 320), U=1024, G=3U=3072, M=B*T=32768.

#define B_   128
#define T_   256
#define E_   300
#define EP_  320
#define U_   1024
#define G_   3072
#define M_   (B_ * T_)

typedef __bf16 bf16_t;
typedef __attribute__((ext_vector_type(16))) __bf16 v16bf;
typedef __attribute__((ext_vector_type(8)))  __bf16 v8bf;
typedef __attribute__((ext_vector_type(8)))  float  v8f;
typedef __attribute__((ext_vector_type(4)))  unsigned int u32x4;
typedef __attribute__((ext_vector_type(8)))  int i32x8;
typedef __attribute__((ext_vector_type(4)))  int i32x4;

__device__ __forceinline__ float sigmoid_(float x) {
    return 1.0f / (1.0f + __expf(-x));
}

// ---------------------------------------------------------------------------
// Tensor Data Mover: async 2D bf16 tile load global -> LDS.
// Descriptor packing per CDNA5 ISA:
//   group0: [1:0]=count=1 | [63:32]=lds_addr | [120:64]=global_addr | [127:126]=type=2
//   group1: [17:16]=data_size=1 (2B) | tensor_dim0/1 | tile_dim0/1 | dim0 stride
// Tracked with TENSORcnt; caller waits via s_wait_tensorcnt then barriers.
// ---------------------------------------------------------------------------
__device__ __forceinline__ void tdm_load_2d_bf16(
    unsigned lds_off,            // LDS byte offset of destination tile
    const void* gptr,            // global byte address of tile start
    unsigned tensor_d0,          // tensor row length (elements)
    unsigned tensor_d1,          // tensor rows
    unsigned tile_d0,            // tile width (elements, contiguous)
    unsigned tile_d1,            // tile height (rows)
    unsigned stride0)            // tensor row stride (elements)
{
    unsigned long long ga = (unsigned long long)gptr;
    u32x4 g0;
    g0[0] = 1u;                                             // count=1, user mode
    g0[1] = lds_off;                                        // lds_addr
    g0[2] = (unsigned)ga;                                   // global_addr[31:0]
    g0[3] = (unsigned)((ga >> 32) & 0x01ffffffull) | (2u << 30);  // addr[56:32]|type=2

    i32x8 g1;
    g1[0] = (int)(1u << 16);                                // wg_mask=0, data_size=2B
    g1[1] = (int)((tensor_d0 & 0xffffu) << 16);             // tdim0[15:0] @ [63:48]
    g1[2] = (int)((tensor_d0 >> 16) | ((tensor_d1 & 0xffffu) << 16));
    g1[3] = (int)((tensor_d1 >> 16) | (tile_d0 << 16));     // tile_dim0 @ [127:112]
    g1[4] = (int)(tile_d1 & 0xffffu);                       // tile_dim1; tile_dim2=0
    g1[5] = (int)stride0;                                   // dim0_stride[31:0]
    g1[6] = 0;                                              // stride hi / dim1_stride
    g1[7] = 0;

    i32x4 z4 = {0, 0, 0, 0};
#if __has_include(<hip/amd_detail/amd_gfx1250_TDM.h>)
    i32x8 z8 = {0, 0, 0, 0, 0, 0, 0, 0};
    __builtin_amdgcn_tensor_load_to_lds(g0, g1, z4, z4, z8, 0);   // clang-23 6-arg
#else
    __builtin_amdgcn_tensor_load_to_lds(g0, g1, z4, z4, 0);       // ROCm 7.2 5-arg
#endif
}

// ---------------------------------------------------------------------------
// Embedding gather + f32 -> bf16, pad E 300 -> 320. One block per (b,t) row.
// ---------------------------------------------------------------------------
__global__ void k_embed(const int* __restrict__ x, const float* __restrict__ emb,
                        bf16_t* __restrict__ out) {
    const int m = blockIdx.x;       // 0..M_-1
    const int e = threadIdx.x;      // 0..319
    const int tok = x[m];
    float v = (e < E_) ? emb[(long)tok * E_ + e] : 0.0f;
    out[(long)m * EP_ + e] = (bf16_t)v;
    if (m + 1 < M_) {
        // warm L2 for the next row's gather (global_prefetch_b8)
        __builtin_prefetch(&emb[(long)x[m + 1] * E_ + e], 0, 1);
    }
}

// ---------------------------------------------------------------------------
// Transpose + convert weight [K,N] f32 -> [N,Kp] bf16 (zero-pad K..Kp).
// ---------------------------------------------------------------------------
__global__ void k_transpose_w(const float* __restrict__ w, bf16_t* __restrict__ wt,
                              int K, int Kp, int N) {
    const int n = blockIdx.x;       // 0..N-1
    for (int k = threadIdx.x; k < Kp; k += blockDim.x) {
        float v = (k < K) ? w[(long)k * N + n] : 0.0f;
        wt[(long)n * Kp + k] = (bf16_t)v;
    }
}

__global__ void k_zero_bf16(bf16_t* __restrict__ p, long n) {
    long i = (long)blockIdx.x * blockDim.x + threadIdx.x;
    if (i < n) p[i] = (bf16_t)0.0f;
}

// ---------------------------------------------------------------------------
// xg = embA[M,320](bf16) x WT[3072,320]^T (bf16) + bias0  -> f32 [M,3072]
// Block tile 128(M) x 64(N), 8 waves, K chunks of 32. A tile via TDM.
// ---------------------------------------------------------------------------
__global__ __launch_bounds__(256) void k_gemm_xg(
    const bf16_t* __restrict__ A, const bf16_t* __restrict__ WT,
    const float* __restrict__ bias0, float* __restrict__ C) {
    __shared__ __align__(16) bf16_t lA[128][32];
    __shared__ __align__(16) bf16_t lB[64][32];

    const int tid  = threadIdx.x;
    const int wave = tid >> 5;
    const int lane = tid & 31;
    const int m0   = blockIdx.x * 128;
    const int n0   = blockIdx.y * 64;
    const unsigned ldsA = (unsigned)(size_t)&lA[0][0];

    v8f acc[4] = {};

    for (int kc = 0; kc < EP_; kc += 32) {
        if (wave == 0) {   // TDM: 128x32 bf16 tile of A, row stride EP_
            tdm_load_2d_bf16(ldsA, A + (long)m0 * EP_ + kc,
                             EP_, M_, 32, 128, EP_);
        }
        {   // stage B tile 64x32 (N-major: WT rows are contiguous in K)
            const int row = tid >> 2;
            const int seg = (tid & 3) * 8;
            *(v8bf*)(&lB[row][seg]) =
                *(const v8bf*)(WT + (long)(n0 + row) * EP_ + kc + seg);
        }
        if (wave == 0) __builtin_amdgcn_s_wait_tensorcnt(0);
        __syncthreads();

        // A fragment (ISA 16-bit A layout: lane<16 -> K {0..7,16..23})
        const int ar = wave * 16 + (lane & 15);
        const int kb = (lane >> 4) * 8;
        v8bf alo = *(const v8bf*)(&lA[ar][kb]);
        v8bf ahi = *(const v8bf*)(&lA[ar][kb + 16]);
        v16bf afrag = __builtin_shufflevector(alo, ahi,
            0,1,2,3,4,5,6,7,8,9,10,11,12,13,14,15);

#pragma unroll
        for (int nt = 0; nt < 4; ++nt) {
            const int bn = nt * 16 + (lane & 15);
            const int bk = (lane >> 4) * 8;
            v8bf blo = *(const v8bf*)(&lB[bn][bk]);
            v8bf bhi = *(const v8bf*)(&lB[bn][bk + 16]);
            v16bf bfrag = __builtin_shufflevector(blo, bhi,
                0,1,2,3,4,5,6,7,8,9,10,11,12,13,14,15);
            acc[nt] = __builtin_amdgcn_wmma_f32_16x16x32_bf16(
                false, afrag, false, bfrag, (short)0, acc[nt], false, false);
        }
        __syncthreads();
    }

#pragma unroll
    for (int nt = 0; nt < 4; ++nt) {
#pragma unroll
        for (int i = 0; i < 8; ++i) {
            const int m = m0 + wave * 16 + i + ((lane >> 4) << 3);
            const int n = n0 + nt * 16 + (lane & 15);
            C[(long)m * G_ + n] = acc[nt][i] + bias0[n];
        }
    }
}

// ---------------------------------------------------------------------------
// One GRU timestep, fully fused: rg = h @ RK + b1 (3 gates via WMMA),
// gates + state update + output write. 16 blocks (64-wide U slices).
// h chunk staged by the Tensor Data Mover; weights via global->LDS b128.
// ---------------------------------------------------------------------------
__global__ __launch_bounds__(256) void k_gru_step(
    const bf16_t* __restrict__ hin,    // [128,1024] bf16
    bf16_t* __restrict__ hout,         // [128,1024] bf16
    const bf16_t* __restrict__ RKT,    // [3072,1024] bf16 (N-major)
    const float* __restrict__ xg,      // [M,3072] f32
    const float* __restrict__ bias1,   // [3072]
    float* __restrict__ out,           // [B,T,U]
    int t) {
    __shared__ __align__(16) bf16_t lA[128][32];
    __shared__ __align__(16) bf16_t lB[3][64][32];

    const int tid  = threadIdx.x;
    const int wave = tid >> 5;
    const int lane = tid & 31;
    const int u0   = blockIdx.x * 64;
    const unsigned ldsA = (unsigned)(size_t)&lA[0][0];

    v8f acc[3][4] = {};

    for (int kc = 0; kc < U_; kc += 32) {
        if (wave == 0) {   // TDM: h chunk 128x32, row stride U_
            tdm_load_2d_bf16(ldsA, hin + kc, U_, B_, 32, 128, U_);
        }
#pragma unroll
        for (int g = 0; g < 3; ++g) {   // 3 gate weight slices, 64x32 each
            const int row = tid >> 2;
            const int seg = (tid & 3) * 8;
            *(v8bf*)(&lB[g][row][seg]) =
                *(const v8bf*)(RKT + (long)(g * U_ + u0 + row) * U_ + kc + seg);
        }
        if (wave == 0) __builtin_amdgcn_s_wait_tensorcnt(0);
        __syncthreads();

        const int ar = wave * 16 + (lane & 15);
        const int kb = (lane >> 4) * 8;
        v8bf alo = *(const v8bf*)(&lA[ar][kb]);
        v8bf ahi = *(const v8bf*)(&lA[ar][kb + 16]);
        v16bf afrag = __builtin_shufflevector(alo, ahi,
            0,1,2,3,4,5,6,7,8,9,10,11,12,13,14,15);

#pragma unroll
        for (int g = 0; g < 3; ++g) {
#pragma unroll
            for (int nt = 0; nt < 4; ++nt) {
                const int bn = nt * 16 + (lane & 15);
                const int bk = (lane >> 4) * 8;
                v8bf blo = *(const v8bf*)(&lB[g][bn][bk]);
                v8bf bhi = *(const v8bf*)(&lB[g][bn][bk + 16]);
                v16bf bfrag = __builtin_shufflevector(blo, bhi,
                    0,1,2,3,4,5,6,7,8,9,10,11,12,13,14,15);
                acc[g][nt] = __builtin_amdgcn_wmma_f32_16x16x32_bf16(
                    false, afrag, false, bfrag, (short)0, acc[g][nt], false, false);
            }
        }
        __syncthreads();
    }

    // Fused gate epilogue (Keras reset_after=True):
    // z = sig(xz+rz); r = sig(xr+rr); hh = tanh(xh + r*rh); h' = z*h + (1-z)*hh
#pragma unroll
    for (int nt = 0; nt < 4; ++nt) {
#pragma unroll
        for (int i = 0; i < 8; ++i) {
            const int b = wave * 16 + i + ((lane >> 4) << 3);   // batch row
            const int u = u0 + nt * 16 + (lane & 15);
            const float* xgp = xg + ((long)b * T_ + t) * G_;
            const float xz = xgp[u];
            const float xr = xgp[U_ + u];
            const float xh = xgp[2 * U_ + u];
            const float rz = acc[0][nt][i] + bias1[u];
            const float rr = acc[1][nt][i] + bias1[U_ + u];
            const float rh = acc[2][nt][i] + bias1[2 * U_ + u];
            const float z  = sigmoid_(xz + rz);
            const float r  = sigmoid_(xr + rr);
            const float hh = tanhf(xh + r * rh);
            const float hp = (float)hin[(long)b * U_ + u];
            const float hn = z * hp + (1.0f - z) * hh;
            out[((long)b * T_ + t) * U_ + u] = hn;
            hout[(long)b * U_ + u] = (bf16_t)hn;
        }
    }
}

__global__ void k_state(const bf16_t* __restrict__ h, float* __restrict__ st) {
    const int i = blockIdx.x * blockDim.x + threadIdx.x;   // B_*U_ elems
    st[i] = (float)h[i];
}

// ---------------------------------------------------------------------------
extern "C" void kernel_launch(void* const* d_in, const int* in_sizes, int n_in,
                              void* d_out, int out_size, void* d_ws, size_t ws_size,
                              hipStream_t stream) {
    const int*   x    = (const int*)d_in[0];
    const float* emb  = (const float*)d_in[1];
    const float* kern = (const float*)d_in[2];   // [300,3072]
    const float* rk   = (const float*)d_in[3];   // [1024,3072]
    const float* bias = (const float*)d_in[4];   // [2,3072]
    float* outp = (float*)d_out;                 // [B,T,U] then [B,U]

    char* ws = (char*)d_ws;
    size_t off = 0;
    auto alloc = [&](size_t bytes) -> void* {
        void* p = ws + off;
        off = (off + bytes + 255) & ~(size_t)255;
        return p;
    };
    bf16_t* embA = (bf16_t*)alloc((size_t)M_ * EP_ * sizeof(bf16_t));
    bf16_t* wT   = (bf16_t*)alloc((size_t)G_ * EP_ * sizeof(bf16_t));
    bf16_t* rkT  = (bf16_t*)alloc((size_t)G_ * U_ * sizeof(bf16_t));
    float*  xg   = (float*)alloc((size_t)M_ * G_ * sizeof(float));
    bf16_t* h0   = (bf16_t*)alloc((size_t)B_ * U_ * sizeof(bf16_t));
    bf16_t* h1   = (bf16_t*)alloc((size_t)B_ * U_ * sizeof(bf16_t));

    // h(t=0) = 0
    k_zero_bf16<<<(B_ * U_ + 255) / 256, 256, 0, stream>>>(h0, (long)B_ * U_);
    // embedding gather -> bf16 padded rows
    k_embed<<<M_, EP_, 0, stream>>>(x, emb, embA);
    // weights -> bf16, N-major
    k_transpose_w<<<G_, 128, 0, stream>>>(kern, wT, E_, EP_, G_);
    k_transpose_w<<<G_, 128, 0, stream>>>(rk, rkT, U_, U_, G_);
    // big input-projection GEMM (done once for all timesteps)
    dim3 gx(M_ / 128, G_ / 64);
    k_gemm_xg<<<gx, 256, 0, stream>>>(embA, wT, bias, xg);
    // sequential recurrent scan: 256 fused step kernels
    bf16_t* hb[2] = {h0, h1};
    for (int t = 0; t < T_; ++t) {
        k_gru_step<<<U_ / 64, 256, 0, stream>>>(
            hb[t & 1], hb[(t + 1) & 1], rkT, xg, bias + G_, outp, t);
    }
    // final state (after 256 steps the live buffer is hb[0])
    k_state<<<(B_ * U_) / 256, 256, 0, stream>>>(
        hb[0], outp + (size_t)B_ * T_ * U_);
}
